// GAT_2860448219497
// MI455X (gfx1250) — compile-verified, matrix operand order checked by reading, not run on hardware
//
#include <hip/hip_runtime.h>
#include <stdint.h>

#define N       4096
#define NF      512
#define HID     64
#define NHEADS  8
#define NOUT    56
#define NOUTP   64
#define ALPHA   0.2f
#define NEGBIG  -9.0e10f

typedef __bf16 bf16_t;
typedef __attribute__((ext_vector_type(16))) __bf16 v16bf;
typedef __attribute__((ext_vector_type(8)))  float  v8f;

union BVec { uint4 u[2]; v16bf v; };

// A/B operand K index for lane/element in the 16x32 bf16 WMMA layout:
// lanes 0-15 hold K {0..7,16..23}, lanes 16-31 hold K {8..15,24..31}
__device__ __forceinline__ int kOf(int lane, int e) {
    return ((e & 8) << 1) + ((lane & 16) >> 1) + (e & 7);
}

// ---------------------------------------------------------------- adj -> bits
__global__ void pack_adj_kernel(const int* __restrict__ adj, uint32_t* __restrict__ adjbits) {
    int tid = blockIdx.x * blockDim.x + threadIdx.x;      // N * 128 words
    int row = tid >> 7;
    int wb  = tid & 127;
    const int4* p = reinterpret_cast<const int4*>(adj + (size_t)row * N + wb * 32);
    uint32_t bits = 0;
#pragma unroll
    for (int q = 0; q < 8; ++q) {
        int4 v = p[q];
        bits |= (uint32_t)(v.x > 0) << (q * 4 + 0);
        bits |= (uint32_t)(v.y > 0) << (q * 4 + 1);
        bits |= (uint32_t)(v.z > 0) << (q * 4 + 2);
        bits |= (uint32_t)(v.w > 0) << (q * 4 + 3);
    }
    adjbits[tid] = bits;
}

// ---------------------------------------------------------------- f32 -> bf16
__global__ void cvt_bf16_kernel(const float* __restrict__ src, bf16_t* __restrict__ dst, int n) {
    int i = blockIdx.x * blockDim.x + threadIdx.x;
    if (i < n) dst[i] = (bf16_t)src[i];
}

// Repack a [heads][rows][64] f32 matrix into the WMMA B-operand lane layout (bf16)
__global__ void pack_B_kernel(const float* __restrict__ src, bf16_t* __restrict__ dst,
                              int rows, int heads) {
    int tid = blockIdx.x * blockDim.x + threadIdx.x;      // heads*rows*64
    int e   = tid & 15;
    int l   = (tid >> 4) & 31;
    int t   = (tid >> 9) & 3;
    int rest = tid >> 11;
    int nkb = rows >> 5;
    int kb  = rest % nkb;
    int h   = rest / nkb;
    int K   = kb * 32 + kOf(l, e);
    int col = t * 16 + (l & 15);
    dst[tid] = (bf16_t)src[((size_t)h * rows + K) * 64 + col];
}

// ---------------------------------------------------------------- H = x @ W  (per head, bf16 WMMA)
__global__ void gemm_heads_kernel(const bf16_t* __restrict__ xb,
                                  const uint4* __restrict__ Wlay,
                                  float* __restrict__ Hf) {
    int w    = threadIdx.x >> 5;
    int lane = threadIdx.x & 31;
    int m0   = blockIdx.x * 128 + w * 16;
    int h    = blockIdx.y;

    v8f acc[4];
#pragma unroll
    for (int t = 0; t < 4; ++t) acc[t] = (v8f)0.0f;

    int rowA = m0 + (lane & 15);
    int hw8  = (lane & 16) ? 8 : 0;

    for (int kb = 0; kb < NF / 32; ++kb) {
        const uint4* pa = reinterpret_cast<const uint4*>(xb + (size_t)rowA * NF + kb * 32 + hw8);
        BVec ua;
        ua.u[0] = pa[0];
        ua.u[1] = pa[2];            // +16 bf16 elements = +2 uint4
        // issue all B-tile loads before consuming anything
        BVec ub[4];
#pragma unroll
        for (int t = 0; t < 4; ++t) {
            const uint4* pb = Wlay + (((size_t)h * (NF / 32) + kb) * 4 + t) * 64 + lane * 2;
            ub[t].u[0] = pb[0]; ub[t].u[1] = pb[1];
        }
#pragma unroll
        for (int t = 0; t < 4; ++t)
            acc[t] = __builtin_amdgcn_wmma_f32_16x16x32_bf16(
                false, ua.v, false, ub[t].v, (short)0, acc[t], false, false);
    }
#pragma unroll
    for (int t = 0; t < 4; ++t) {
#pragma unroll
        for (int r = 0; r < 8; ++r) {
            int ro  = m0 + r + ((lane >> 4) << 3);
            int col = t * 16 + (lane & 15);
            Hf[((size_t)h * N + ro) * HID + col] = acc[t][r];
        }
    }
}

// ---------------------------------------------------------------- s1/s2 per (head,node)
__global__ void s1s2_kernel(const float* __restrict__ Hf, const float* __restrict__ As,
                            float* __restrict__ s1, float* __restrict__ s2) {
    int tid = blockIdx.x * blockDim.x + threadIdx.x;      // NHEADS*N
    int h = tid >> 12, i = tid & (N - 1);
    float a = 0.f, b = 0.f;
    const float* hrow = Hf + ((size_t)h * N + i) * HID;
#pragma unroll 8
    for (int c = 0; c < HID; ++c) {
        float v = hrow[c];
        a += v * As[h * 128 + c];
        b += v * As[h * 128 + 64 + c];
    }
    s1[tid] = a; s2[tid] = b;
}

// ------------------------- fused masked softmax + att@H (flash style, WMMA) + ELU
__global__ void gat_attn_kernel(const float* __restrict__ s1, const float* __restrict__ s2,
                                const uint32_t* __restrict__ adjbits,
                                const uint4* __restrict__ Blay,
                                float* __restrict__ out,
                                int sHeadStride, int bHeadStrideU4, int outHeadStride) {
    __shared__ float s2sh[N];
    int h = blockIdx.y;
    const float* s2h = s2 + (size_t)h * sHeadStride;
    for (int i = threadIdx.x; i < N; i += 256) s2sh[i] = s2h[i];
    __syncthreads();

    int w    = threadIdx.x >> 5;
    int lane = threadIdx.x & 31;
    int m0   = blockIdx.x * 128 + w * 16;
    int row  = m0 + (lane & 15);
    float s1r = s1[(size_t)h * sHeadStride + row];
    int hi8  = (lane >> 4) << 3;              // D-layout row offset for this lane half
    int h8   = (lane & 16) ? 8 : 0;           // A-layout K-half for this lane

    float m = -3.0e38f, lsum = 0.f;
    v8f acc[4];
#pragma unroll
    for (int t = 0; t < 4; ++t) acc[t] = (v8f)0.0f;

    const uint4* bh = Blay + (size_t)h * bHeadStrideU4 + lane * 2;

    for (int jb = 0; jb < N / 32; ++jb) {
        // --- issue all global loads for this step up front (adj word + 4 B tiles)
        uint32_t aw = adjbits[(size_t)row * (N / 32) + jb];
        BVec ub[4];
#pragma unroll
        for (int t = 0; t < 4; ++t) {
            const uint4* pb = bh + ((size_t)(jb * 4 + t)) * 64;
            ub[t].u[0] = pb[0]; ub[t].u[1] = pb[1];
        }

        // --- scores: this lane's 16 s2 values are two contiguous float8 runs
        int j0 = jb * 32;
        const float4* q0 = reinterpret_cast<const float4*>(s2sh + j0 + h8);
        const float4* q1 = reinterpret_cast<const float4*>(s2sh + j0 + 16 + h8);
        float4 sA = q0[0], sB = q0[1], sC = q1[0], sD = q1[1];
        float sv[16] = { sA.x, sA.y, sA.z, sA.w, sB.x, sB.y, sB.z, sB.w,
                         sC.x, sC.y, sC.z, sC.w, sD.x, sD.y, sD.z, sD.w };

        float ev[16];
        float tmax = -3.0e38f;
#pragma unroll
        for (int e = 0; e < 16; ++e) {
            int kk  = ((e & 8) << 1) + h8 + (e & 7);
            float x = s1r + sv[e];
            x = fmaxf(x, ALPHA * x);                      // leaky relu (slope < 1)
            x = ((aw >> kk) & 1u) ? x : NEGBIG;           // adjacency mask
            ev[e] = x;
            tmax = fmaxf(tmax, x);
        }
        tmax = fmaxf(tmax, __shfl_xor(tmax, 16, 32));     // combine both K-halves of each row

        // --- flash update; rescale only if any row's max grew (wave-uniform branch)
        float mnew = m;
        if (__ballot(tmax > m)) {
            mnew = fmaxf(m, tmax);
            float scale = __expf(m - mnew);
            lsum *= scale;                                // per-lane stats are per-row
            float scv[8];
#pragma unroll
            for (int r = 0; r < 8; ++r) scv[r] = __shfl(scale, r + hi8, 32);
#pragma unroll
            for (int t = 0; t < 4; ++t)
#pragma unroll
                for (int r = 0; r < 8; ++r) acc[t][r] *= scv[r];
        }

        float psum = 0.f;
        v16bf pa;
#pragma unroll
        for (int e = 0; e < 16; ++e) {
            float p = __expf(ev[e] - mnew);
            psum += p;
            pa[e] = (bf16_t)p;
        }
        psum += __shfl_xor(psum, 16, 32);
        lsum += psum;

#pragma unroll
        for (int t = 0; t < 4; ++t)
            acc[t] = __builtin_amdgcn_wmma_f32_16x16x32_bf16(
                false, pa, false, ub[t].v, (short)0, acc[t], false, false);
        m = mnew;
    }

    float lv[8];
#pragma unroll
    for (int r = 0; r < 8; ++r) lv[r] = __shfl(lsum, r + hi8, 32);
#pragma unroll
    for (int t = 0; t < 4; ++t) {
#pragma unroll
        for (int r = 0; r < 8; ++r) {
            float v = acc[t][r] / lv[r];
            v = v > 0.f ? v : (__expf(v) - 1.f);          // ELU
            int ro = m0 + r + hi8;
            out[(size_t)h * outHeadStride + (size_t)ro * 64 + t * 16 + (lane & 15)] = v;
        }
    }
}

// ---------------------------------------------------------------- z = mean over heads
__global__ void reduce_z_kernel(const float* __restrict__ Hp, float* __restrict__ z) {
    int tid = blockIdx.x * blockDim.x + threadIdx.x;      // N*HID
    float s = 0.f;
#pragma unroll
    for (int h = 0; h < NHEADS; ++h) s += Hp[(size_t)h * N * HID + tid];
    z[tid] = s * (1.0f / NHEADS);
}

// ---------------------------------------------------------------- h2 = z @ W_out (padded to 64)
__global__ void h2_kernel(const float* __restrict__ z, const float* __restrict__ Wo,
                          float* __restrict__ h2) {
    int tid = blockIdx.x * blockDim.x + threadIdx.x;      // N*NOUTP
    int i = tid >> 6, c = tid & 63;
    float s = 0.f;
    if (c < NOUT) {
        const float* zr = z + (size_t)i * HID;
#pragma unroll 8
        for (int k = 0; k < HID; ++k) s += zr[k] * Wo[k * NOUT + c];
    }
    h2[tid] = s;                                          // zero pad cols 56..63
}

// ---------------------------------------------------------------- output-layer s1/s2
__global__ void s1s2_out_kernel(const float* __restrict__ h2, const float* __restrict__ ao,
                                float* __restrict__ s1o, float* __restrict__ s2o) {
    int i = blockIdx.x * blockDim.x + threadIdx.x;        // N
    float a = 0.f, b = 0.f;
    const float* hr = h2 + (size_t)i * NOUTP;
#pragma unroll 8
    for (int c = 0; c < NOUT; ++c) {
        float v = hr[c];
        a += v * ao[c];
        b += v * ao[NOUT + c];
    }
    s1o[i] = a; s2o[i] = b;
}

// ---------------------------------------------------------------- final row softmax (width 56)
__global__ void softmax_kernel(const float* __restrict__ h2p, float* __restrict__ out) {
    int i = blockIdx.x * blockDim.x + threadIdx.x;        // N
    const float* r = h2p + (size_t)i * NOUTP;
    float mx = -3.0e38f;
#pragma unroll
    for (int c = 0; c < NOUT; ++c) mx = fmaxf(mx, r[c]);
    float s = 0.f;
    float ex[NOUT];
#pragma unroll
    for (int c = 0; c < NOUT; ++c) { ex[c] = __expf(r[c] - mx); s += ex[c]; }
    float inv = 1.0f / s;
#pragma unroll
    for (int c = 0; c < NOUT; ++c) out[(size_t)i * NOUT + c] = ex[c] * inv;
}

extern "C" void kernel_launch(void* const* d_in, const int* in_sizes, int n_in,
                              void* d_out, int out_size, void* d_ws, size_t ws_size,
                              hipStream_t stream) {
    (void)in_sizes; (void)n_in; (void)out_size; (void)ws_size;
    const float* x   = (const float*)d_in[0];
    const int*   adj = (const int*)d_in[1];
    const float* Ws  = (const float*)d_in[2];
    const float* As  = (const float*)d_in[3];
    const float* Wo  = (const float*)d_in[4];
    const float* ao  = (const float*)d_in[5];
    float* out = (float*)d_out;

    char* ws = (char*)d_ws;
    size_t off = 0;
    auto take = [&](size_t bytes) -> char* {
        char* p = ws + off;
        off += (bytes + 255) & ~(size_t)255;
        return p;
    };
    uint32_t* adjbits = (uint32_t*)take((size_t)N * (N / 32) * 4);          // 2 MB
    bf16_t*   xb      = (bf16_t*)  take((size_t)N * NF * 2);                // 4 MB
    bf16_t*   Wlay    = (bf16_t*)  take((size_t)NHEADS * NF * HID * 2);     // 512 KB
    float*    Hf      = (float*)   take((size_t)NHEADS * N * HID * 4);      // 8 MB
    bf16_t*   Hlay    = (bf16_t*)  take((size_t)NHEADS * N * HID * 2);      // 4 MB
    float*    s1      = (float*)   take((size_t)NHEADS * N * 4);
    float*    s2      = (float*)   take((size_t)NHEADS * N * 4);
    float*    Hp      = (float*)   take((size_t)NHEADS * N * HID * 4);      // 8 MB
    float*    z       = (float*)   take((size_t)N * HID * 4);               // 1 MB
    float*    h2      = (float*)   take((size_t)N * NOUTP * 4);             // 1 MB
    bf16_t*   h2lay   = (bf16_t*)  take((size_t)N * NOUTP * 2);             // 512 KB
    float*    s1o     = (float*)   take((size_t)N * 4);
    float*    s2o     = (float*)   take((size_t)N * 4);
    float*    h2p     = (float*)   take((size_t)N * NOUTP * 4);             // 1 MB

    // 1. adjacency bitmask (only full 64 MB read)
    pack_adj_kernel<<<(N * (N / 32)) / 256, 256, 0, stream>>>(adj, adjbits);
    // 2. x -> bf16
    cvt_bf16_kernel<<<(N * NF) / 256, 256, 0, stream>>>(x, xb, N * NF);
    // 3. Ws -> WMMA-B layout
    pack_B_kernel<<<(NHEADS * NF * HID) / 256, 256, 0, stream>>>(Ws, Wlay, NF, NHEADS);
    // 4. H = x @ W per head  (WMMA bf16)
    gemm_heads_kernel<<<dim3(N / 128, NHEADS), 256, 0, stream>>>(xb, (const uint4*)Wlay, Hf);
    // 5. attention logits s1, s2
    s1s2_kernel<<<(NHEADS * N) / 256, 256, 0, stream>>>(Hf, As, s1, s2);
    // 6. H -> WMMA-B layout for att@H
    pack_B_kernel<<<(NHEADS * N * HID) / 256, 256, 0, stream>>>(Hf, Hlay, N, NHEADS);
    // 7. fused masked softmax + att@H + ELU per head (WMMA bf16, flash streaming)
    gat_attn_kernel<<<dim3(N / 128, NHEADS), 256, 0, stream>>>(
        s1, s2, adjbits, (const uint4*)Hlay, Hp, N, N * HID / 8, N * HID);
    // 8. z = mean over heads
    reduce_z_kernel<<<(N * HID) / 256, 256, 0, stream>>>(Hp, z);
    // 9. h2 = z @ W_out (zero-padded to 64 cols)
    h2_kernel<<<(N * NOUTP) / 256, 256, 0, stream>>>(z, Wo, h2);
    // 10. output-layer logits
    s1s2_out_kernel<<<N / 256, 256, 0, stream>>>(h2, ao, s1o, s2o);
    // 11. h2 -> WMMA-B layout
    pack_B_kernel<<<(N * NOUTP) / 256, 256, 0, stream>>>(h2, h2lay, N, 1);
    // 12. output-layer attention (+ELU, reference applies elu after concat=False layer)
    gat_attn_kernel<<<dim3(N / 128, 1), 256, 0, stream>>>(
        s1o, s2o, adjbits, (const uint4*)h2lay, h2p, 0, 0, 0);
    // 13. final row softmax over 56 classes
    softmax_kernel<<<N / 256, 256, 0, stream>>>(h2p, out);
}
